// AttnPointNetConv_63891933495552
// MI455X (gfx1250) — matrix operand order repaired and checked by reading, not compiled, not run on hardware
//
#include <hip/hip_runtime.h>
#include <cstdint>

typedef __attribute__((ext_vector_type(16))) _Float16 v16h;
typedef __attribute__((ext_vector_type(8)))  _Float16 v8h;
typedef __attribute__((ext_vector_type(8)))  float    v8f;

#define BN_EPS 1e-5f

__device__ __forceinline__ void wave_sync_lds() {
  asm volatile("s_wait_dscnt 0" ::: "memory");
}

__device__ __forceinline__ float silu_f(float v) { return v / (1.f + __expf(-v)); }

__device__ __forceinline__ unsigned pkh(float a, float b) {
  union { _Float16 h[2]; unsigned u; } z;
  z.h[0] = (_Float16)a; z.h[1] = (_Float16)b; return z.u;
}

// order-preserving float<->uint encode for atomicMax on signed floats
__device__ __forceinline__ unsigned fenc(float f) {
  unsigned u = __float_as_uint(f);
  return (u & 0x80000000u) ? ~u : (u | 0x80000000u);
}
__device__ __forceinline__ float fdec(unsigned u) {
  return (u & 0x80000000u) ? __uint_as_float(u & 0x7FFFFFFFu) : __uint_as_float(~u);
}

// ---------------------------------------------------------------- init
__global__ void k_init(float* out, unsigned* gmax, float* denom,
                       float* stats1, float* gstats, int N, long long total) {
  long long i = (long long)blockIdx.x * blockDim.x + threadIdx.x;
  if (i < total) out[i] = 0.f;
  if (i < N) { gmax[i] = 0u; denom[i] = 0.f; }
  if (i < 128) stats1[i] = 0.f;
  if (i < 2) gstats[i] = 0.f;
}

// ------------------------------------------------- W1^T (f16, padded 96)
// K-reorder: k<64 -> x part (row k+3 of W1), k=64..66 -> pos part, rest 0
__global__ void k_prepW(const float* __restrict__ W1, _Float16* __restrict__ W1T) {
  int idx = blockIdx.x * blockDim.x + threadIdx.x;
  if (idx >= 64 * 96) return;
  int n = idx / 96, k = idx % 96;
  float w = 0.f;
  if (k < 64)      w = W1[(k + 3) * 64 + n];
  else if (k < 67) w = W1[(k - 64) * 64 + n];
  W1T[idx] = (_Float16)w;
}

// ------------------------------------------- message GEMM via WMMA f16
// wave: 16 edges x 64 channels, K=96 (3 x K32 tiles), 12 WMMAs / tile
__global__ void __launch_bounds__(256)
k_msg_gemm(const float* __restrict__ x, const float* __restrict__ pos,
           const float* __restrict__ b1, const _Float16* __restrict__ W1T,
           const int* __restrict__ srcI, const int* __restrict__ dstI,
           _Float16* __restrict__ hpre, float* __restrict__ stats1, int ntiles) {
  __shared__ __align__(16) _Float16 Wbuf[64][96];
  __shared__ __align__(16) _Float16 Abuf[8][16][96];
  const int tid = threadIdx.x, lane = tid & 31, wv = tid >> 5;

  for (int i = tid; i < 64 * 96; i += 256) Wbuf[i / 96][i % 96] = W1T[i];
  __syncthreads();

  // resident B fragments: N=lane%16, K 0..15 / 16..31 by lane half (this code's contiguous order, same as A)
  v16h Bf[12];
  {
    const int row = (lane & 15);
    const int kh = (lane & 16) ? 16 : 0;
#pragma unroll
    for (int nt = 0; nt < 4; ++nt)
#pragma unroll
      for (int t = 0; t < 3; ++t) {
        v8h lo = *(const v8h*)&Wbuf[nt * 16 + row][t * 32 + kh];
        v8h hi = *(const v8h*)&Wbuf[nt * 16 + row][t * 32 + kh + 8];
        Bf[nt * 3 + t] = __builtin_shufflevector(
            lo, hi, 0, 1, 2, 3, 4, 5, 6, 7, 8, 9, 10, 11, 12, 13, 14, 15);
      }
  }

  float b1v[4];
#pragma unroll
  for (int nt = 0; nt < 4; ++nt) b1v[nt] = b1[nt * 16 + (lane & 15)];

  // zero K-pad (cols 68..95) of this wave's A staging buffer, once
  for (int i = lane; i < 16 * 28; i += 32)
    Abuf[wv][i / 28][68 + i % 28] = (_Float16)0.f;

  float sAcc[4] = {0, 0, 0, 0}, qAcc[4] = {0, 0, 0, 0};

  const int nw = gridDim.x * 8;
  for (int tile = blockIdx.x * 8 + wv; tile < ntiles; tile += nw) {
    const int base = tile * 16;
    const int r = lane >> 1, hf = lane & 1;
    const long long e = (long long)base + r;
    const int sN = srcI[e], dN = dstI[e];

    wave_sync_lds();   // prior-iter LDS reads done before restaging
    {  // gather + f16-pack one half-row of the message matrix
      const float4* xp = (const float4*)(x + (long long)sN * 64 + hf * 32);
      unsigned pk[16];
#pragma unroll
      for (int j = 0; j < 8; ++j) {
        float4 v = xp[j];
        pk[2 * j] = pkh(v.x, v.y);
        pk[2 * j + 1] = pkh(v.z, v.w);
      }
      uint4* ab = (uint4*)&Abuf[wv][r][hf * 32];
#pragma unroll
      for (int j = 0; j < 4; ++j)
        ab[j] = make_uint4(pk[4 * j], pk[4 * j + 1], pk[4 * j + 2], pk[4 * j + 3]);
      if (hf == 0) {
        float p0 = pos[(long long)sN * 3 + 0] - pos[(long long)dN * 3 + 0];
        float p1 = pos[(long long)sN * 3 + 1] - pos[(long long)dN * 3 + 1];
        float p2 = pos[(long long)sN * 3 + 2] - pos[(long long)dN * 3 + 2];
        uint2 pu; pu.x = pkh(p0, p1); pu.y = pkh(p2, 0.f);
        *(uint2*)&Abuf[wv][r][64] = pu;
      }
    }
    wave_sync_lds();

    v8f acc[4];
#pragma unroll
    for (int nt = 0; nt < 4; ++nt)
#pragma unroll
      for (int i = 0; i < 8; ++i) acc[nt][i] = b1v[nt];

    {  // A fragments: this code's contiguous k order (same as B): K 0..15 lanes<16, 16..31 lanes>=16
      const int row = lane & 15;
      const int koff = (lane & 16) ? 16 : 0;
#pragma unroll
      for (int t = 0; t < 3; ++t) {
        v8h lo = *(const v8h*)&Abuf[wv][row][t * 32 + koff];
        v8h hi = *(const v8h*)&Abuf[wv][row][t * 32 + koff + 8];
        v16h Af = __builtin_shufflevector(
            lo, hi, 0, 1, 2, 3, 4, 5, 6, 7, 8, 9, 10, 11, 12, 13, 14, 15);
#pragma unroll
        for (int nt = 0; nt < 4; ++nt)
          acc[nt] = __builtin_amdgcn_wmma_f32_16x16x32_f16(
              false, Af, false, Bf[nt * 3 + t], (short)0, acc[nt], false, false);
      }
    }

    // BN stats + transpose h tile (f16) through LDS for coalesced stores
    wave_sync_lds();
    const int mo = (lane & 16) ? 8 : 0;
#pragma unroll
    for (int nt = 0; nt < 4; ++nt) {
      const int n = nt * 16 + (lane & 15);
      float s_ = 0.f, q_ = 0.f;
#pragma unroll
      for (int i = 0; i < 8; ++i) {
        float v = acc[nt][i];
        s_ += v; q_ += v * v;
        Abuf[wv][mo + i][n] = (_Float16)v;
      }
      sAcc[nt] += s_; qAcc[nt] += q_;
    }
    wave_sync_lds();
    {
      const int m = lane >> 1;
      const uint4* sp = (const uint4*)&Abuf[wv][m][hf * 32];
      uint4* gp = (uint4*)(hpre + (long long)(base + m) * 64 + hf * 32);
#pragma unroll
      for (int j = 0; j < 4; ++j) gp[j] = sp[j];
    }
  }

  // lanes (L, L+16) share a channel: pairwise reduce, then global atomics
#pragma unroll
  for (int nt = 0; nt < 4; ++nt) {
    float s_ = sAcc[nt] + __shfl_xor(sAcc[nt], 16, 32);
    float q_ = qAcc[nt] + __shfl_xor(qAcc[nt], 16, 32);
    if (lane < 16) {
      atomicAdd(&stats1[nt * 16 + lane], s_);
      atomicAdd(&stats1[64 + nt * 16 + lane], q_);
    }
  }
}

// -------------------------------------- scalar tail (E%16 edges, if any)
__global__ void k_msg_tail(const float* x, const float* pos, const float* W1,
                           const float* b1, const int* srcI, const int* dstI,
                           _Float16* hpre, float* stats1, int e0, int E) {
  int idx = blockIdx.x * blockDim.x + threadIdx.x;
  int rem = E - e0;
  if (idx >= rem * 64) return;
  int e = e0 + idx / 64, ch = idx % 64;
  int sN = srcI[e], dN = dstI[e];
  float h = b1[ch];
  for (int i = 0; i < 3; ++i)
    h += (pos[(long long)sN * 3 + i] - pos[(long long)dN * 3 + i]) * W1[i * 64 + ch];
  for (int j = 0; j < 64; ++j)
    h += x[(long long)sN * 64 + j] * W1[(3 + j) * 64 + ch];
  hpre[(long long)e * 64 + ch] = (_Float16)h;
  atomicAdd(&stats1[ch], h);
  atomicAdd(&stats1[64 + ch], h * h);
}

// ---------------------------------------------------------- BN1 finalize
__global__ void k_bn1(const float* stats1, const float* g, const float* b,
                      float* bn1c, float invE) {
  int c = threadIdx.x;
  if (c >= 64) return;
  float mean = stats1[c] * invE;
  float var = fmaxf(stats1[64 + c] * invE - mean * mean, 0.f);
  float sc = g[c] * rsqrtf(var + BN_EPS);
  bn1c[c] = sc;
  bn1c[64 + c] = b[c] - mean * sc;
}

// ---------------------------------- gate pre-activation + its BN stats
__global__ void __launch_bounds__(256)
k_gate_pre(const unsigned* __restrict__ hpre32, const float* __restrict__ bn1c,
           const float* __restrict__ Wg, const float* __restrict__ bg,
           float* __restrict__ gate_pre, float* __restrict__ gstats, int E) {
  const int lane = threadIdx.x & 31;
  const int gw = blockIdx.x * (blockDim.x >> 5) + (threadIdx.x >> 5);
  const int nw = gridDim.x * (blockDim.x >> 5);
  const int c0 = 2 * lane;
  const float sc0 = bn1c[c0], sc1 = bn1c[c0 + 1];
  const float sh0 = bn1c[64 + c0], sh1 = bn1c[64 + c0 + 1];
  const float w0 = Wg[c0], w1 = Wg[c0 + 1];
  const float bgv = bg[0];
  float ls = 0.f, lq = 0.f;
  for (long long e = gw; e < E; e += nw) {
    union { unsigned u; _Float16 h[2]; } z;
    z.u = hpre32[e * 32 + lane];
    float h0 = silu_f(fmaf((float)z.h[0], sc0, sh0));
    float h1 = silu_f(fmaf((float)z.h[1], sc1, sh1));
    float p = h0 * w0 + h1 * w1;
#pragma unroll
    for (int o = 16; o >= 1; o >>= 1) p += __shfl_down(p, o, 32);
    if (lane == 0) {
      float gp = p + bgv;
      gate_pre[e] = gp;
      ls += gp; lq += gp * gp;
    }
  }
  if (lane == 0) { atomicAdd(&gstats[0], ls); atomicAdd(&gstats[1], lq); }
}

__global__ void k_bng(const float* gstats, const float* g, const float* b,
                      float* gc, float invE) {
  if (threadIdx.x != 0 || blockIdx.x != 0) return;
  float mean = gstats[0] * invE;
  float var = fmaxf(gstats[1] * invE - mean * mean, 0.f);
  float sc = g[0] * rsqrtf(var + BN_EPS);
  gc[0] = sc; gc[1] = b[0] - mean * sc;
}

// -------------------------------------------------- gate + segment max
__global__ void k_gate_max(const float* gate_pre, const float* gc,
                           const int* dstI, float* gate, unsigned* gmax, int E) {
  long long e = (long long)blockIdx.x * blockDim.x + threadIdx.x;
  if (e >= E) return;
  float gv = silu_f(fmaf(gate_pre[e], gc[0], gc[1]));
  gate[e] = gv;
  atomicMax(&gmax[dstI[e]], fenc(gv));
}

// ----------------------------------------------- exp + segment denom
__global__ void k_exp_den(const float* gate, const unsigned* gmax,
                          const int* dstI, float* ex, float* denom, int E) {
  long long e = (long long)blockIdx.x * blockDim.x + threadIdx.x;
  if (e >= E) return;
  int d = dstI[e];
  float v = __expf(gate[e] - fdec(gmax[d]));
  ex[e] = v;
  atomicAdd(&denom[d], v);
}

// --------------------------- attention-weighted scatter (recompute h)
__global__ void __launch_bounds__(256)
k_scatter(const unsigned* __restrict__ hpre32, const float* __restrict__ bn1c,
          const float* __restrict__ ex, const float* __restrict__ denom,
          const int* __restrict__ dstI, float* __restrict__ out, int E) {
  const int lane = threadIdx.x & 31;
  const int gw = blockIdx.x * (blockDim.x >> 5) + (threadIdx.x >> 5);
  const int nw = gridDim.x * (blockDim.x >> 5);
  const int c0 = 2 * lane;
  const float sc0 = bn1c[c0], sc1 = bn1c[c0 + 1];
  const float sh0 = bn1c[64 + c0], sh1 = bn1c[64 + c0 + 1];
  for (long long e = gw; e < E; e += nw) {
    const int d = dstI[e];
    const float a = ex[e] / fmaxf(denom[d], 1e-16f);
    union { unsigned u; _Float16 h[2]; } z;
    z.u = hpre32[e * 32 + lane];
    float h0 = silu_f(fmaf((float)z.h[0], sc0, sh0));
    float h1 = silu_f(fmaf((float)z.h[1], sc1, sh1));
    atomicAdd(&out[(long long)d * 64 + c0], h0 * a);
    atomicAdd(&out[(long long)d * 64 + c0 + 1], h1 * a);
  }
}

extern "C" void kernel_launch(void* const* d_in, const int* in_sizes, int n_in,
                              void* d_out, int out_size, void* d_ws, size_t ws_size,
                              hipStream_t stream) {
  const float* x    = (const float*)d_in[0];
  const float* pos  = (const float*)d_in[1];
  const float* W1   = (const float*)d_in[2];
  const float* b1   = (const float*)d_in[3];
  const float* bn1g = (const float*)d_in[4];
  const float* bn1b = (const float*)d_in[5];
  const float* Wg   = (const float*)d_in[6];
  const float* bg   = (const float*)d_in[7];
  const float* bngg = (const float*)d_in[8];
  const float* bngb = (const float*)d_in[9];
  const int*   eidx = (const int*)d_in[10];

  const int N = in_sizes[0] / 64;
  const int E = in_sizes[10] / 2;
  const int* srcI = eidx;
  const int* dstI = eidx + E;

  char* w = (char*)d_ws;
  auto carve = [&](size_t bytes) -> char* {
    char* p = w; w += (bytes + 255) & ~(size_t)255; return p;
  };
  _Float16* hpre  = (_Float16*)carve((size_t)E * 64 * 2);
  float* gate_pre = (float*)carve((size_t)E * 4);
  float* gate     = (float*)carve((size_t)E * 4);
  float* exb      = (float*)carve((size_t)E * 4);
  unsigned* gmax  = (unsigned*)carve((size_t)N * 4);
  float* denom    = (float*)carve((size_t)N * 4);
  float* stats1   = (float*)carve(128 * 4);
  float* bn1c     = (float*)carve(128 * 4);
  float* gstats   = (float*)carve(8 * 4);
  float* gc       = (float*)carve(8 * 4);
  _Float16* W1T   = (_Float16*)carve(64 * 96 * 2);

  float* out = (float*)d_out;
  const long long total = (long long)N * 64;
  const float invE = 1.f / (float)E;

  k_init<<<(unsigned)((total + 255) / 256), 256, 0, stream>>>(
      out, gmax, denom, stats1, gstats, N, total);
  k_prepW<<<(64 * 96 + 255) / 256, 256, 0, stream>>>(W1, W1T);

  const int ntiles = E / 16;
  k_msg_gemm<<<1024, 256, 0, stream>>>(x, pos, b1, W1T, srcI, dstI, hpre,
                                       stats1, ntiles);
  const int e0 = ntiles * 16;
  if (e0 < E) {
    int rem = E - e0;
    k_msg_tail<<<(rem * 64 + 255) / 256, 256, 0, stream>>>(
        x, pos, W1, b1, srcI, dstI, hpre, stats1, e0, E);
  }
  k_bn1<<<1, 64, 0, stream>>>(stats1, bn1g, bn1b, bn1c, invE);
  k_gate_pre<<<2048, 256, 0, stream>>>((const unsigned*)hpre, bn1c, Wg, bg,
                                       gate_pre, gstats, E);
  k_bng<<<1, 1, 0, stream>>>(gstats, bngg, bngb, gc, invE);
  k_gate_max<<<(E + 255) / 256, 256, 0, stream>>>(gate_pre, gc, dstI, gate,
                                                  gmax, E);
  k_exp_den<<<(E + 255) / 256, 256, 0, stream>>>(gate, gmax, dstI, exb, denom, E);
  k_scatter<<<8192, 256, 0, stream>>>((const unsigned*)hpre, bn1c, exb, denom,
                                      dstI, out, E);
}
